// Native_Attention_msa_onlyAttn_4664334483727
// MI455X (gfx1250) — matrix operand order repaired and checked by reading, not compiled, not run on hardware
//
#include <hip/hip_runtime.h>
#include <hip/hip_bf16.h>

typedef __attribute__((ext_vector_type(16))) _Float16 v16h;
typedef __attribute__((ext_vector_type(8)))  _Float16 v8h;
typedef __attribute__((ext_vector_type(4)))  _Float16 v4h;
typedef __attribute__((ext_vector_type(8)))  float    v8f;

#define B_SZ   2
#define N_TOK  4096
#define C_DIM  256
#define H_NUM  8
#define D_HEAD 32
#define SCALE  25.0f
#define SSTRIDE 4100   // 4096 + 4 floats pad (keeps cross-row LDS accesses off one bank)

union V16U { v16h v; v8h h[2]; };

// ---------------- kernel 0a: x f32 -> f16 (4 elems/thread) ----------------
__global__ __launch_bounds__(256) void cvt_x_kernel(const float* __restrict__ x,
                                                    _Float16* __restrict__ xh) {
    int i = blockIdx.x * 256 + threadIdx.x;           // 524288 threads, 4 floats each
    float4 v = ((const float4*)x)[i];
    v4h o;
    o[0] = (_Float16)v.x; o[1] = (_Float16)v.y;
    o[2] = (_Float16)v.z; o[3] = (_Float16)v.w;
    ((v4h*)xh)[i] = o;
}

// ---------------- kernel 0b: W [256x512] f32 -> Wt [512x256] f16 ----------------
__global__ __launch_bounds__(256) void cvt_wt_kernel(const float* __restrict__ w,
                                                     _Float16* __restrict__ wt) {
    int idx = blockIdx.x * 256 + threadIdx.x;         // 131072 threads
    int j = idx >> 8;          // output column of W (0..511)
    int i = idx & 255;         // input row (0..255)
    wt[j * C_DIM + i] = (_Float16)w[i * (2 * C_DIM) + j];
}

// ---------------- kernel 1: qkv = x @ W  (WMMA f16, f32 accum) ----------------
// one wave per 16x16 output tile; 512 M-tiles x 32 N-tiles = 16384 waves
__global__ __launch_bounds__(256) void qkv_gemm_kernel(const _Float16* __restrict__ xh,
                                                       const _Float16* __restrict__ wt,
                                                       float* __restrict__ qkv) {
    const int tid  = threadIdx.x;
    const int lane = tid & 31;
    const int wave = tid >> 5;
    const int tile = blockIdx.x * 8 + wave;           // 0..16383
    const int mt = tile >> 5;                         // 0..511
    const int nt = tile & 31;                         // 0..31

    const int arow = lane & 15;
    const int ahs  = lane >> 4;

    const _Float16* aprow = xh + (size_t)(mt * 16 + arow) * C_DIM;
    const _Float16* bprow = wt + (size_t)(nt * 16 + arow) * C_DIM + ahs * 16;

    v8f c = {};
#pragma unroll
    for (int kk = 0; kk < 8; ++kk) {                  // K = 8 * 32 = 256
        V16U a, b;
        const _Float16* ap = aprow + kk * 32 + ahs * 16;
        a.h[0] = *(const v8h*)(ap);                   // K = {0..7}  (+16*hs)
        a.h[1] = *(const v8h*)(ap + 8);               // K = {8..15} (+16*hs)
        const _Float16* bp = bprow + kk * 32;
        b.h[0] = *(const v8h*)(bp);                   // contiguous 32B of column
        b.h[1] = *(const v8h*)(bp + 8);
        c = __builtin_amdgcn_wmma_f32_16x16x32_f16(false, a.v, false, b.v,
                                                   (short)0, c, false, false);
    }
    // D layout: lanes 0-15 -> rows r, lanes 16-31 -> rows r+8
    float* obase = qkv + (size_t)(mt * 16) * (2 * C_DIM) + nt * 16 + (lane & 15);
    const int rbase = ahs * 8;
#pragma unroll
    for (int r = 0; r < 8; ++r)
        obase[(size_t)(rbase + r) * (2 * C_DIM)] = c[r];
}

// ---- kernel 2: L2-normalize rows, emit q (pre-scaled by 25) / k f16 [B,H,N,d] ----
__global__ __launch_bounds__(256) void norm_kernel(const float* __restrict__ qkv,
                                                   _Float16* __restrict__ qh,
                                                   _Float16* __restrict__ kh) {
    int rid  = blockIdx.x * 256 + threadIdx.x;        // 131072 rows of 32
    int ng   = rid >> 4;                              // token 0..8191
    int rest = rid & 15;
    int t    = rest >> 3;                             // 0=q, 1=k
    int h    = rest & 7;
    const float* src = qkv + (size_t)ng * (2 * C_DIM) + t * C_DIM + h * D_HEAD;
    float buf[D_HEAD];
    float ss = 0.f;
#pragma unroll
    for (int c = 0; c < D_HEAD; ++c) { float v = src[c]; buf[c] = v; ss += v * v; }
    float rn = 1.0f / (sqrtf(ss) + 1e-8f);
    if (t == 0) rn *= SCALE;                          // fold softmax scale into q
    int b = ng >> 12, nn = ng & (N_TOK - 1);
    _Float16* dst = (t ? kh : qh) + ((size_t)((b * H_NUM + h) * N_TOK + nn)) * D_HEAD;
#pragma unroll
    for (int c = 0; c < D_HEAD; ++c) dst[c] = (_Float16)(buf[c] * rn);
}

// ---------------- kernel 3: scores (WMMA) + row softmax in LDS ----------------
// one WG per (b,h, 16-query tile): 16 * 256 = 4096 WGs, 8 waves each.
__global__ __launch_bounds__(256) void attn_kernel(const _Float16* __restrict__ qh,
                                                   const _Float16* __restrict__ kh,
                                                   float* __restrict__ out) {
    extern __shared__ float sc[];                     // 16*SSTRIDE scores + 256 red
    float* red = sc + 16 * SSTRIDE;

    const int tid  = threadIdx.x;
    const int lane = tid & 31;
    const int wave = tid >> 5;
    const int bh   = blockIdx.x >> 8;                 // 0..15
    const int qt   = blockIdx.x & 255;                // 0..255

    const _Float16* qbase = qh + ((size_t)bh * N_TOK + qt * 16) * D_HEAD;
    const _Float16* kbase = kh + (size_t)bh * N_TOK * D_HEAD;

    // A operand: 16x32 q tile (already scaled by 25)
    const int arow = lane & 15;
    const int ahs  = lane >> 4;
    V16U a;
    const _Float16* ap = qbase + arow * D_HEAD + ahs * 16;
    a.h[0] = *(const v8h*)(ap);
    a.h[1] = *(const v8h*)(ap + 8);

    // each wave covers 512 keys = 32 key tiles; one WMMA per tile (K = d = 32)
    for (int kt = 0; kt < 32; ++kt) {
        const int key0 = wave * 512 + kt * 16;
        const _Float16* kp = kbase + (size_t)(key0 + (lane & 15)) * D_HEAD + (lane >> 4) * 16;
        if (kt + 1 < 32) __builtin_prefetch(kp + 16 * D_HEAD, 0, 1);  // global_prefetch_b8
        V16U b;
        b.h[0] = *(const v8h*)(kp);
        b.h[1] = *(const v8h*)(kp + 8);
        v8f c = {};
        c = __builtin_amdgcn_wmma_f32_16x16x32_f16(false, a.v, false, b.v,
                                                   (short)0, c, false, false);
        const int col   = key0 + (lane & 15);
        const int rbase = (lane >> 4) * 8;
#pragma unroll
        for (int r = 0; r < 8; ++r)
            sc[(rbase + r) * SSTRIDE + col] = c[r];
    }
    __syncthreads();

    // softmax: thread t owns row = t>>4, float4-columns c0 + 16*j (interleaved);
    // per store instruction a wave writes two contiguous 256B bursts.
    const int row = tid >> 4;
    const int c0  = tid & 15;
    float4* rp4 = (float4*)(sc + row * SSTRIDE);

    float lmax = -3.4e38f;
#pragma unroll 4
    for (int j = 0; j < 64; ++j) {
        float4 v = rp4[c0 + 16 * j];
        lmax = fmaxf(lmax, fmaxf(fmaxf(v.x, v.y), fmaxf(v.z, v.w)));
    }
    red[tid] = lmax;                                  // tid == row*16 + c0
    __syncthreads();
    float m = -3.4e38f;
#pragma unroll
    for (int s = 0; s < 16; ++s) m = fmaxf(m, red[row * 16 + s]);
    __syncthreads();

    float lsum = 0.f;
#pragma unroll 4
    for (int j = 0; j < 64; ++j) {
        float4 v = rp4[c0 + 16 * j];
        v.x = __expf(v.x - m);                        // v_exp_f32
        v.y = __expf(v.y - m);
        v.z = __expf(v.z - m);
        v.w = __expf(v.w - m);
        rp4[c0 + 16 * j] = v;
        lsum += v.x + v.y + v.z + v.w;
    }
    red[tid] = lsum;
    __syncthreads();
    float tot = 0.f;
#pragma unroll
    for (int s = 0; s < 16; ++s) tot += red[row * 16 + s];
    float rinv = 1.0f / tot;

    float4* op = (float4*)(out + ((size_t)blockIdx.x * 16 + row) * N_TOK);
#pragma unroll 4
    for (int j = 0; j < 64; ++j) {
        float4 v = rp4[c0 + 16 * j];
        v.x *= rinv; v.y *= rinv; v.z *= rinv; v.w *= rinv;
        op[c0 + 16 * j] = v;                          // wave: 2x 256B contiguous
    }
}

// ---------------- launch ----------------
extern "C" void kernel_launch(void* const* d_in, const int* in_sizes, int n_in,
                              void* d_out, int out_size, void* d_ws, size_t ws_size,
                              hipStream_t stream) {
    const float* x = (const float*)d_in[0];   // [2,4096,256]
    const float* w = (const float*)d_in[1];   // [256,512]
    float* out = (float*)d_out;               // [2,8,4096,4096]

    char* ws = (char*)d_ws;
    _Float16* xh  = (_Float16*)(ws);                         //  4.00 MB
    _Float16* wt  = (_Float16*)(ws +  4u * 1024 * 1024);     //  0.25 MB
    float*    qkv = (float*)   (ws +  5u * 1024 * 1024);     // 16.00 MB
    _Float16* qh  = (_Float16*)(ws + 22u * 1024 * 1024);     //  4.00 MB
    _Float16* kh  = (_Float16*)(ws + 27u * 1024 * 1024);     //  4.00 MB

    cvt_x_kernel   <<<2048, 256, 0, stream>>>(x, xh);
    cvt_wt_kernel  <<< 512, 256, 0, stream>>>(w, wt);
    qkv_gemm_kernel<<<2048, 256, 0, stream>>>(xh, wt, qkv);
    norm_kernel    <<< 512, 256, 0, stream>>>(qkv, qh, kh);
    attn_kernel    <<<4096, 256, (16 * SSTRIDE + 256) * sizeof(float), stream>>>(qh, kh, out);
}
